// MambaClassification_15762529976996
// MI455X (gfx1250) — matrix operand order repaired and checked by reading, not compiled, not run on hardware
//
#include <hip/hip_runtime.h>
#include <hip/hip_bf16.h>

// ---------------- model dims ----------------
#define BB      2
#define LL      2048
#define BLR     (BB*LL)          // 4096 rows
#define IN_SZ   256
#define DM      1024
#define NCLS    128
#define NLAY    2
#define ED      2048
#define NST     16
#define DTR     64
#define DCONV   4
#define EPSV    1e-5f

typedef __bf16 bf16_t;
typedef __attribute__((ext_vector_type(16))) __bf16 v16bf;
typedef __attribute__((ext_vector_type(8)))  __bf16 v8bf;
typedef __attribute__((ext_vector_type(8)))  float  v8f;

union ABu { v16bf v; v8bf h[2]; };

__device__ __forceinline__ unsigned short f2bf_rne(float f) {
    unsigned u = __float_as_uint(f);
    unsigned r = u + 0x7FFFu + ((u >> 16) & 1u);
    return (unsigned short)(r >> 16);
}

// ---------------- elementwise: f32 -> bf16 ----------------
__global__ void k_cvt_bf16(const float* __restrict__ src,
                           unsigned short* __restrict__ dst, size_t n) {
    size_t i = (size_t)blockIdx.x * blockDim.x + threadIdx.x;
    if (i < n) dst[i] = f2bf_rne(src[i]);
}

// ---------------- rmsnorm (fp32 in, bf16 out), one block per row ----------------
__global__ void k_rmsnorm_bf16(const float* __restrict__ h,
                               const float* __restrict__ w,
                               unsigned short* __restrict__ out, int D) {
    __shared__ float red[256];
    int row = blockIdx.x;
    const float* hr = h + (size_t)row * D;
    float ss = 0.f;
    for (int i = threadIdx.x; i < D; i += 256) { float v = hr[i]; ss += v * v; }
    red[threadIdx.x] = ss; __syncthreads();
    for (int off = 128; off > 0; off >>= 1) {
        if (threadIdx.x < off) red[threadIdx.x] += red[threadIdx.x + off];
        __syncthreads();
    }
    float scale = rsqrtf(red[0] / (float)D + EPSV);
    unsigned short* orow = out + (size_t)row * D;
    for (int i = threadIdx.x; i < D; i += 256)
        orow[i] = f2bf_rne(hr[i] * scale * w[i]);
}

// ---------------- bf16 WMMA GEMM: C = A[M,K] * W[N,K]^T (+bias) (+=C) ----------------
// Wave computes a (16*MT) x (16*NT) strip. Double buffering is written with
// separately-named register buffers and a manual unroll-by-2 over K (K % 64 == 0)
// so no runtime-indexed arrays exist (keeps everything in VGPRs, no spills).
template <int MT, int NT>
__launch_bounds__(256, 1)
__global__ void k_gemm_bf16(const bf16_t* __restrict__ A, const bf16_t* __restrict__ W,
                            const float* __restrict__ bias, float* __restrict__ C,
                            unsigned short* __restrict__ Cb, int accum,
                            int M, int N, int K, int lda, int ldw, int ldc) {
    int wave = threadIdx.x >> 5;
    int lane = threadIdx.x & 31;
    int tilesN = N / (16 * NT);
    int tilesM = M / (16 * MT);
    int tileId = blockIdx.x * 8 + wave;
    if (tileId >= tilesM * tilesN) return;    // wave-uniform: EXEC stays all-ones
    int mt = tileId / tilesN;
    int nt = tileId % tilesN;
    int m0 = mt * 16 * MT, n0 = nt * 16 * NT;
    int ln = lane & 15;                       // A-row (M) / B-col (N) index within tile
    int half = lane >> 4;

    // Lane-invariant K packing:
    //  A tile 16x32: lane holds K = [k+half*8, +8) and [k+16+half*8, +8)
    //  B tile 32x16: lane holds column n, K = [k+half*8, +8) and [k+16+half*8, +8)
    const bf16_t* aBase = A + (size_t)(m0 + ln) * lda + half * 8;
    const bf16_t* wBase = W + (size_t)(n0 + ln) * ldw + half * 8;

    v8f acc[MT][NT];
#pragma unroll
    for (int mi = 0; mi < MT; ++mi)
#pragma unroll
        for (int ti = 0; ti < NT; ++ti)
#pragma unroll
            for (int i = 0; i < 8; ++i) acc[mi][ti][i] = 0.f;

    ABu a0[MT], a1[MT], b0[NT], b1[NT];

    auto loadA = [&](ABu* dst, int k) {
#pragma unroll
        for (int mi = 0; mi < MT; ++mi) {
            const bf16_t* ap = aBase + (size_t)mi * 16 * lda + k;
            dst[mi].h[0] = *(const v8bf*)(ap);
            dst[mi].h[1] = *(const v8bf*)(ap + 16);
        }
    };
    auto loadB = [&](ABu* dst, int k) {
#pragma unroll
        for (int ti = 0; ti < NT; ++ti) {
            const bf16_t* bp = wBase + (size_t)ti * 16 * ldw + k;
            dst[ti].h[0] = *(const v8bf*)(bp);
            dst[ti].h[1] = *(const v8bf*)(bp + 16);
        }
    };

    loadA(a0, 0);
    loadB(b0, 0);
    for (int k = 0; k < K; k += 64) {
        // stage next half-step while buf0 computes
        loadA(a1, k + 32);
        loadB(b1, k + 32);
#pragma unroll
        for (int mi = 0; mi < MT; ++mi)
#pragma unroll
            for (int ti = 0; ti < NT; ++ti)
                acc[mi][ti] = __builtin_amdgcn_wmma_f32_16x16x32_bf16(
                    false, a0[mi].v, false, b0[ti].v, (short)0, acc[mi][ti], false, false);
        if (k + 64 < K) {
            loadA(a0, k + 64);
            loadB(b0, k + 64);
        }
#pragma unroll
        for (int mi = 0; mi < MT; ++mi)
#pragma unroll
            for (int ti = 0; ti < NT; ++ti)
                acc[mi][ti] = __builtin_amdgcn_wmma_f32_16x16x32_bf16(
                    false, a1[mi].v, false, b1[ti].v, (short)0, acc[mi][ti], false, false);
    }

    // C layout: VGPR v -> row (tile_m0 + v + 8*half), column (tile_n0 + ln)
#pragma unroll
    for (int mi = 0; mi < MT; ++mi)
#pragma unroll
        for (int ti = 0; ti < NT; ++ti) {
            int cn = n0 + ti * 16 + ln;
            float bv = bias ? bias[cn] : 0.f;
#pragma unroll
            for (int v = 0; v < 8; ++v) {
                int row = m0 + mi * 16 + v + 8 * half;
                size_t idx = (size_t)row * ldc + cn;
                float val = acc[mi][ti][v] + bv;
                if (accum) val += C[idx];
                C[idx] = val;
                if (Cb) Cb[idx] = f2bf_rne(val);
            }
        }
}

// ---------------- depthwise causal conv(4) + bias + silu ----------------
__global__ void k_conv_silu(const float* __restrict__ xz, const float* __restrict__ cw,
                            const float* __restrict__ cb, float* __restrict__ xc,
                            unsigned short* __restrict__ xcb) {
    size_t idx = (size_t)blockIdx.x * blockDim.x + threadIdx.x;
    if (idx >= (size_t)BLR * ED) return;
    int e = (int)(idx % ED);
    size_t r = idx / ED;
    int l = (int)(r % LL);
    const float* base = xz + r * (size_t)(2 * ED) + e;   // x half of xz, stride 2*ED
    float acc = cb[e];
#pragma unroll
    for (int j = 0; j < DCONV; ++j) {
        int ll = l - (DCONV - 1) + j;
        if (ll >= 0) acc += cw[e * DCONV + j] * base[(long)(ll - l) * (2 * ED)];
    }
    float s = acc / (1.f + __expf(-acc));   // silu
    xc[idx] = s;
    xcb[idx] = f2bf_rne(s);
}

// ---------------- selective scan (softplus fused), sequential over L ----------------
// blockDim = 64; grid = B * (ED/64).
__global__ void k_scan(const float* __restrict__ dpre, const float* __restrict__ dtb,
                       const float* __restrict__ xc, const float* __restrict__ dbc,
                       const float* __restrict__ A_log, float* __restrict__ ys) {
    __shared__ float sB[NST], sC[NST];
    int b = blockIdx.x >> 5;                        // 32 blocks of 64 ch per batch
    int e = ((blockIdx.x & 31) << 6) + threadIdx.x;
    float A[NST], h[NST];
#pragma unroll
    for (int n = 0; n < NST; ++n) {
        A[n] = -__expf(A_log[(size_t)e * NST + n]);
        h[n] = 0.f;
    }
    float dbias = dtb[e];
    for (int l = 0; l < LL; ++l) {
        size_t r = (size_t)b * LL + l;
        __syncthreads();
        if (threadIdx.x < 2 * NST) {
            float v = dbc[r * (DTR + 2 * NST) + DTR + threadIdx.x];
            if (threadIdx.x < NST) sB[threadIdx.x] = v;
            else                   sC[threadIdx.x - NST] = v;
        }
        __syncthreads();
        float dp = dpre[r * ED + e] + dbias;
        float d = (dp > 20.f) ? dp : log1pf(__expf(dp));  // softplus
        float x = xc[r * ED + e];
        float dx = d * x, y = 0.f;
#pragma unroll
        for (int n = 0; n < NST; ++n) {
            h[n] = __expf(d * A[n]) * h[n] + dx * sB[n];
            y += h[n] * sC[n];
        }
        ys[r * ED + e] = y;
    }
}

// ---------------- y = (ys + D*xc) * silu(z), emit bf16 ----------------
__global__ void k_ygate(const float* __restrict__ ys, const float* __restrict__ xc,
                        const float* __restrict__ xz, const float* __restrict__ Dp,
                        unsigned short* __restrict__ yb) {
    size_t idx = (size_t)blockIdx.x * blockDim.x + threadIdx.x;
    if (idx >= (size_t)BLR * ED) return;
    int e = (int)(idx % ED);
    size_t r = idx / ED;
    float z = xz[r * (size_t)(2 * ED) + ED + e];
    float sz = z / (1.f + __expf(-z));
    float y = (ys[idx] + Dp[e] * xc[idx]) * sz;
    yb[idx] = f2bf_rne(y);
}

// ---------------- host side ----------------
static inline void run_gemm(int NT, const unsigned short* A, const unsigned short* W,
                            const float* bias, float* C, unsigned short* Cb, int accum,
                            int M, int N, int K, int lda, int ldw, int ldc,
                            hipStream_t s) {
    int tilesN = N / (16 * NT);
    int tilesM = M / 32;                          // MT = 2
    int waves = tilesM * tilesN;
    int blocks = (waves + 7) / 8;
    if (NT == 4)
        k_gemm_bf16<2, 4><<<blocks, 256, 0, s>>>((const bf16_t*)A, (const bf16_t*)W, bias,
                                                 C, Cb, accum, M, N, K, lda, ldw, ldc);
    else
        k_gemm_bf16<2, 2><<<blocks, 256, 0, s>>>((const bf16_t*)A, (const bf16_t*)W, bias,
                                                 C, Cb, accum, M, N, K, lda, ldw, ldc);
}

static inline void run_cvt(const float* src, unsigned short* dst, size_t n, hipStream_t s) {
    k_cvt_bf16<<<(unsigned)((n + 255) / 256), 256, 0, s>>>(src, dst, n);
}

extern "C" void kernel_launch(void* const* d_in, const int* in_sizes, int n_in,
                              void* d_out, int out_size, void* d_ws, size_t ws_size,
                              hipStream_t stream) {
    const float* x         = (const float*)d_in[0];
    const float* in_head_w = (const float*)d_in[1];
    const float* in_head_b = (const float*)d_in[2];
    const float* cls_w     = (const float*)d_in[3];
    const float* cls_b     = (const float*)d_in[4];
    const float* in_proj_w = (const float*)d_in[5];
    const float* conv_w    = (const float*)d_in[6];
    const float* conv_b    = (const float*)d_in[7];
    const float* x_proj_w  = (const float*)d_in[8];
    const float* dt_proj_w = (const float*)d_in[9];
    const float* dt_proj_b = (const float*)d_in[10];
    const float* A_log     = (const float*)d_in[11];
    const float* D_param   = (const float*)d_in[12];
    const float* out_proj_w= (const float*)d_in[13];
    const float* norm_w    = (const float*)d_in[14];
    float* out = (float*)d_out;

    // ---- carve workspace ----
    char* ws = (char*)d_ws;
    size_t off = 0;
    auto carve = [&](size_t bytes) -> void* {
        void* p = ws + off;
        off += (bytes + 255) & ~(size_t)255;
        return p;
    };
    unsigned short* wb_inhead = (unsigned short*)carve((size_t)DM * IN_SZ * 2);
    unsigned short* wb_inproj = (unsigned short*)carve((size_t)NLAY * 2 * ED * DM * 2);
    unsigned short* wb_xproj  = (unsigned short*)carve((size_t)NLAY * (DTR + 2 * NST) * ED * 2);
    unsigned short* wb_dtproj = (unsigned short*)carve((size_t)NLAY * ED * DTR * 2);
    unsigned short* wb_outproj= (unsigned short*)carve((size_t)NLAY * DM * ED * 2);
    unsigned short* wb_cls    = (unsigned short*)carve((size_t)NCLS * DM * 2);
    unsigned short* xb        = (unsigned short*)carve((size_t)BLR * IN_SZ * 2);
    float*          hbuf      = (float*)carve((size_t)BLR * DM * 4);
    unsigned short* ub        = (unsigned short*)carve((size_t)BLR * DM * 2);
    float*          xzbuf     = (float*)carve((size_t)BLR * 2 * ED * 4);
    float*          xcbuf     = (float*)carve((size_t)BLR * ED * 4);
    unsigned short* xcb       = (unsigned short*)carve((size_t)BLR * ED * 2);
    float*          dbcbuf    = (float*)carve((size_t)BLR * (DTR + 2 * NST) * 4);
    unsigned short* dbcb      = (unsigned short*)carve((size_t)BLR * (DTR + 2 * NST) * 2);
    float*          dlt       = (float*)carve((size_t)BLR * ED * 4);   // delta pre-act
    float*          ysbuf     = (float*)carve((size_t)BLR * ED * 4);
    unsigned short* yb        = (unsigned short*)carve((size_t)BLR * ED * 2);
    (void)ws_size; (void)in_sizes; (void)n_in; (void)out_size;

    // ---- one-time bf16 conversion of weights + input ----
    run_cvt(in_head_w, wb_inhead, (size_t)DM * IN_SZ, stream);
    run_cvt(in_proj_w, wb_inproj, (size_t)NLAY * 2 * ED * DM, stream);
    run_cvt(x_proj_w,  wb_xproj,  (size_t)NLAY * (DTR + 2 * NST) * ED, stream);
    run_cvt(dt_proj_w, wb_dtproj, (size_t)NLAY * ED * DTR, stream);
    run_cvt(out_proj_w,wb_outproj,(size_t)NLAY * DM * ED, stream);
    run_cvt(cls_w,     wb_cls,    (size_t)NCLS * DM, stream);
    run_cvt(x,         xb,        (size_t)BLR * IN_SZ, stream);

    // ---- input head: h = x @ in_head_w^T + b ----
    run_gemm(4, xb, wb_inhead, in_head_b, hbuf, nullptr, 0,
             BLR, DM, IN_SZ, IN_SZ, IN_SZ, DM, stream);

    const size_t nED = (size_t)BLR * ED;
    for (int l = 0; l < NLAY; ++l) {
        // rmsnorm(h) -> u (bf16)
        k_rmsnorm_bf16<<<BLR, 256, 0, stream>>>(hbuf, norm_w + (size_t)l * DM, ub, DM);
        // xz = u @ in_proj^T
        run_gemm(4, ub, wb_inproj + (size_t)l * 2 * ED * DM, nullptr, xzbuf, nullptr, 0,
                 BLR, 2 * ED, DM, DM, DM, 2 * ED, stream);
        // causal depthwise conv + silu (fp32 + bf16 outputs)
        k_conv_silu<<<(unsigned)((nED + 255) / 256), 256, 0, stream>>>(
            xzbuf, conv_w + (size_t)l * ED * DCONV, conv_b + (size_t)l * ED, xcbuf, xcb);
        // dbc = xconv @ x_proj^T (N = 96), bf16 sidecar feeds dt GEMM
        run_gemm(2, xcb, wb_xproj + (size_t)l * (DTR + 2 * NST) * ED, nullptr,
                 dbcbuf, dbcb, 0, BLR, DTR + 2 * NST, ED, ED, ED, DTR + 2 * NST, stream);
        // dpre = dt @ dt_proj^T  (softplus deferred into scan)
        run_gemm(4, dbcb, wb_dtproj + (size_t)l * ED * DTR, nullptr, dlt, nullptr, 0,
                 BLR, ED, DTR, DTR + 2 * NST, DTR, ED, stream);
        // selective scan (softplus + bias fused)
        k_scan<<<BB * (ED / 64), 64, 0, stream>>>(dlt, dt_proj_b + (size_t)l * ED,
                                                  xcbuf, dbcbuf,
                                                  A_log + (size_t)l * ED * NST, ysbuf);
        // y = (ys + D*x) * silu(z) -> bf16
        k_ygate<<<(unsigned)((nED + 255) / 256), 256, 0, stream>>>(
            ysbuf, xcbuf, xzbuf, D_param + (size_t)l * ED, yb);
        // h += y @ out_proj^T  (accumulate in GEMM; last layer also emits bf16 h)
        run_gemm(4, yb, wb_outproj + (size_t)l * DM * ED, nullptr, hbuf,
                 (l == NLAY - 1) ? ub : nullptr, 1,
                 BLR, DM, ED, ED, ED, DM, stream);
    }

    // ---- class head: out = h @ cls_w^T + b ----
    run_gemm(4, ub, wb_cls, cls_b, out, nullptr, 0,
             BLR, NCLS, DM, DM, DM, NCLS, stream);
}
